// Net_16372415332675
// MI455X (gfx1250) — matrix operand order repaired and checked by reading, not compiled, not run on hardware
//
#include <hip/hip_runtime.h>

// ---------------- CDNA5 WMMA types ----------------
typedef __attribute__((ext_vector_type(16))) __bf16 v16bf;
typedef __attribute__((ext_vector_type(8)))  float  v8f;

#define N_NODES 100000
#define N_EDGES 1600000
#define D_IN    100
#define D_HID   50
#define N_REL   237
#define N_REL2  474
#define N_BASES 5
#define N_CLASS 50
// Y columns: 5*50 basis projections + 50 root projection + 4 pad = 19 WMMA N-tiles
#define YCOLS   304
#define NTILES  (YCOLS / 16)

__device__ __forceinline__ unsigned short f2bf(float f) {
    unsigned int u = __float_as_uint(f);
    u += 0x7FFFu + ((u >> 16) & 1u);          // round-to-nearest-even
    return (unsigned short)(u >> 16);
}
__device__ __forceinline__ float bf2f(unsigned short s) {
    return __uint_as_float(((unsigned int)s) << 16);
}

// ---------------- A: f32 [M,K] -> bf16 [M,Kp] zero-padded, target row zeroed ----------------
__global__ void a2bf_kernel(const float* __restrict__ A, unsigned short* __restrict__ Abf,
                            int K, int Kp, int total, const int* __restrict__ tgtPtr)
{
    const int t = blockIdx.x * 256 + threadIdx.x;
    if (t >= total) return;
    const int n = t / Kp, k = t - n * Kp;
    const int tgt = tgtPtr ? tgtPtr[0] : -1;
    const float v = (k < K && n != tgt) ? A[(size_t)n * K + k] : 0.f;
    Abf[t] = f2bf(v);
}

// ---------------- B: pack [basis0..4 | root | pad] directly into WMMA fragment order ----------
// Bsw[(nt*KT + kb)*32 + lane][16] ; lane<16 -> K = kb*32+0..15, lane>=16 -> K = kb*32+16..31,
// col = nt*16 + (lane&15)   (contiguous k order per lane half, same as gemm_wmma's A loads)
__global__ __launch_bounds__(32) void bcat_sw_kernel(const float* __restrict__ basis,
                                                     const float* __restrict__ root,
                                                     unsigned short* __restrict__ Bsw,
                                                     int K, int KT)
{
    const int kb   = blockIdx.x % KT;
    const int nt   = blockIdx.x / KT;
    const int lane = threadIdx.x;
    const int col  = nt * 16 + (lane & 15);
    const int kbase = kb * 32 + (lane >> 4) * 16;
    unsigned short o[16];
    #pragma unroll
    for (int j = 0; j < 16; ++j) {
        const int k = kbase + j;
        float v = 0.f;
        if (k < K && col < N_BASES * 50 + 50) {
            v = (col < N_BASES * 50)
                    ? basis[(size_t)(col / 50) * K * 50 + k * 50 + (col % 50)]
                    : root[k * 50 + (col - N_BASES * 50)];
        }
        o[j] = f2bf(v);
    }
    uint4* d = (uint4*)(Bsw + ((size_t)blockIdx.x * 32 + lane) * 16);
    d[0] = *(const uint4*)(o);
    d[1] = *(const uint4*)(o + 8);
}

// ---------------- dense GEMM via v_wmma_f32_16x16x32_bf16 ----------------
// Y[M, YCOLS] (bf16) = Abf[M, KT*32] (bf16) @ B (pre-swizzled fragments)
// one wave per 16x16 tile; M = 100000 = 6250*16 exactly; KT fully unrolled.
template <int KT>
__global__ __launch_bounds__(32) void gemm_wmma(
    const unsigned short* __restrict__ Abf,
    const unsigned short* __restrict__ Bsw,
    unsigned short* __restrict__ Y)
{
    const int lane = threadIdx.x;
    const int row0 = blockIdx.x * 16;
    const int nt   = blockIdx.y;
    const int rlo  = lane & 15;
    const int hi   = lane >> 4;

    // A 16x32 bf16 fragment: lane<16 -> K base..base+15, base += 16 for hi lanes
    const unsigned short* __restrict__ ap = Abf + (size_t)(row0 + rlo) * (KT * 32);
    const unsigned short* __restrict__ bp = Bsw + ((size_t)nt * KT * 32 + lane) * 16;

    v8f c = {0.f, 0.f, 0.f, 0.f, 0.f, 0.f, 0.f, 0.f};

    #pragma unroll
    for (int kb = 0; kb < KT; ++kb) {
        union Frag { v16bf v; uint4 q[2]; } a, b;
        const int abase = kb * 32 + hi * 16;
        a.q[0] = *(const uint4*)(ap + abase);          // K base..base+7
        a.q[1] = *(const uint4*)(ap + abase + 8);      // K base+8..base+15
        b.q[0] = *(const uint4*)(bp + kb * 512);       // 32B contiguous fragment
        b.q[1] = *(const uint4*)(bp + kb * 512 + 8);
        c = __builtin_amdgcn_wmma_f32_16x16x32_bf16(false, a.v, false, b.v,
                                                    (short)0, c, false, false);
    }
    // C/D: lane col = lane&15, VGPR r -> row = row0 + hi*8 + r
    const int mbase = row0 + hi * 8;
    const int bcol  = nt * 16 + rlo;
    #pragma unroll
    for (int r = 0; r < 8; ++r)
        Y[(size_t)(mbase + r) * YCOLS + bcol] = f2bf(c[r]);
}

// ---------------- tiny relation-level precomputes ----------------
__global__ __launch_bounds__(64) void relbias1_kernel(
    const float* __restrict__ rel_emb, const float* __restrict__ comp,
    const float* __restrict__ basis, float* __restrict__ relbias)
{
    const int r = blockIdx.x, j = threadIdx.x;
    if (j >= D_HID) return;
    const float* er = rel_emb + (size_t)(r < N_REL ? r : r - N_REL) * D_IN;
    float acc = 0.f;
    #pragma unroll
    for (int b = 0; b < N_BASES; ++b) {
        const float* bb = basis + (size_t)b * D_IN * D_HID;
        float tb = 0.f;
        for (int i = 0; i < D_IN; ++i) tb += er[i] * bb[i * D_HID + j];
        acc += comp[r * N_BASES + b] * tb;
    }
    relbias[r * D_HID + j] = acc;
}

__global__ __launch_bounds__(64) void e2_kernel(
    const float* __restrict__ rel_emb, const float* __restrict__ relw,
    float* __restrict__ e2)
{
    const int r = blockIdx.x, j = threadIdx.x;
    if (j >= D_HID) return;
    const float* er = rel_emb + (size_t)(r < N_REL ? r : r - N_REL) * D_IN;
    float acc = 0.f;
    for (int i = 0; i < D_IN; ++i) acc += er[i] * relw[i * D_HID + j];
    e2[r * D_HID + j] = acc;
}

__global__ __launch_bounds__(64) void relbias2_kernel(
    const float* __restrict__ e2, const float* __restrict__ comp,
    const float* __restrict__ basis, float* __restrict__ relbias)
{
    const int r = blockIdx.x, j = threadIdx.x;
    if (j >= N_CLASS) return;
    float acc = 0.f;
    #pragma unroll
    for (int b = 0; b < N_BASES; ++b) {
        const float* bb = basis + (size_t)b * D_HID * N_CLASS;
        float tb = 0.f;
        for (int i = 0; i < D_HID; ++i) tb += e2[r * D_HID + i] * bb[i * N_CLASS + j];
        acc += comp[r * N_BASES + b] * tb;
    }
    relbias[r * N_CLASS + j] = acc;
}

// ---------------- graph kernels ----------------
__global__ void count_kernel(const int* __restrict__ dst, float* __restrict__ cnt)
{
    const int e = blockIdx.x * 256 + threadIdx.x;
    if (e >= N_EDGES) return;
    atomicAdd(&cnt[dst[e]], 1.f);
}

// per (edge, j): acc = relbias[type][j] + sum_b comp[type,b]*Y[src][b*50+j]; agg[dst][j] += acc
__global__ __launch_bounds__(256) void edge_kernel(
    const unsigned short* __restrict__ Y,
    const int* __restrict__ src, const int* __restrict__ dst,
    const int* __restrict__ etype,
    const float* __restrict__ comp, const float* __restrict__ relbias,
    float* __restrict__ agg)
{
    const int t = blockIdx.x * 256 + threadIdx.x;
    const int e = t >> 6;
    const int j = t & 63;
    if (e >= N_EDGES || j >= 50) return;
    const int s = src[e], d = dst[e], ty = etype[e];
    const unsigned short* __restrict__ y = Y + (size_t)s * YCOLS;
    float acc = relbias[ty * 50 + j];
    #pragma unroll
    for (int b = 0; b < N_BASES; ++b)
        acc += comp[ty * N_BASES + b] * bf2f(y[b * 50 + j]);
    atomicAdd(&agg[(size_t)d * 50 + j], acc);
}

// h = agg/max(cnt,1) + Y[:,250:300] (root proj) + bias
__global__ void h_kernel(const float* __restrict__ agg, const float* __restrict__ cnt,
                         const unsigned short* __restrict__ Y,
                         const float* __restrict__ bias, float* __restrict__ h)
{
    const int t = blockIdx.x * 256 + threadIdx.x;
    if (t >= N_NODES * 50) return;
    const int n = t / 50, j = t - n * 50;
    const float c = fmaxf(cnt[n], 1.f);
    h[t] = agg[t] / c + bf2f(Y[(size_t)n * YCOLS + N_BASES * 50 + j]) + bias[j];
}

// logits -> log_softmax, one wave (32 lanes) per node, lane covers j and j+32
__global__ __launch_bounds__(256) void final_kernel(
    const float* __restrict__ agg, const float* __restrict__ cnt,
    const unsigned short* __restrict__ Y, const float* __restrict__ bias,
    float* __restrict__ out)
{
    const int node = blockIdx.x * 8 + (threadIdx.x >> 5);
    const int lane = threadIdx.x & 31;
    const float c = fmaxf(cnt[node], 1.f);
    float v0 = -INFINITY, v1 = -INFINITY;
    if (lane < N_CLASS)
        v0 = agg[(size_t)node * 50 + lane] / c +
             bf2f(Y[(size_t)node * YCOLS + N_BASES * 50 + lane]) + bias[lane];
    if (lane + 32 < N_CLASS)
        v1 = agg[(size_t)node * 50 + lane + 32] / c +
             bf2f(Y[(size_t)node * YCOLS + N_BASES * 50 + lane + 32]) + bias[lane + 32];
    float m = fmaxf(v0, v1);
    #pragma unroll
    for (int off = 16; off; off >>= 1) m = fmaxf(m, __shfl_xor(m, off, 32));
    float s = 0.f;
    if (lane < N_CLASS)      s += __expf(v0 - m);
    if (lane + 32 < N_CLASS) s += __expf(v1 - m);
    #pragma unroll
    for (int off = 16; off; off >>= 1) s += __shfl_xor(s, off, 32);
    const float ls = __logf(s);
    if (lane < N_CLASS)      out[(size_t)node * 50 + lane]      = v0 - m - ls;
    if (lane + 32 < N_CLASS) out[(size_t)node * 50 + lane + 32] = v1 - m - ls;
}

// ---------------- host orchestration ----------------
extern "C" void kernel_launch(void* const* d_in, const int* in_sizes, int n_in,
                              void* d_out, int out_size, void* d_ws, size_t ws_size,
                              hipStream_t stream)
{
    const float* x        = (const float*)d_in[0];
    const int*   eidx     = (const int*)d_in[1];
    const int*   etype    = (const int*)d_in[2];
    const int*   tgt      = (const int*)d_in[3];
    const float* rel_emb  = (const float*)d_in[4];
    const float* relw     = (const float*)d_in[5];
    const float* basis1   = (const float*)d_in[6];
    const float* comp1    = (const float*)d_in[7];
    const float* root1    = (const float*)d_in[8];
    const float* bias1    = (const float*)d_in[9];
    const float* basis2   = (const float*)d_in[10];
    const float* comp2    = (const float*)d_in[11];
    const float* root2    = (const float*)d_in[12];
    const float* bias2    = (const float*)d_in[13];
    float* out = (float*)d_out;

    const int* src = eidx;
    const int* dst = eidx + N_EDGES;

    // workspace carve-up (all offsets 256B aligned)
    char* ws = (char*)d_ws;
    size_t off = 0;
    unsigned short* Y   = (unsigned short*)(ws + off); off += (size_t)N_NODES * YCOLS * 2;  // 60.8 MB
    unsigned short* Abf = (unsigned short*)(ws + off); off += (size_t)N_NODES * 128 * 2;    // 25.6 MB
    float* agg          = (float*)(ws + off);          off += (size_t)N_NODES * 50 * 4;     // 20 MB
    float* h            = (float*)(ws + off);          off += (size_t)N_NODES * 50 * 4;     // 20 MB
    float* cnt          = (float*)(ws + off);          off += ((size_t)N_NODES * 4 + 255) & ~(size_t)255;
    unsigned short* Bsw = (unsigned short*)(ws + off); off += (size_t)NTILES * 4 * 32 * 16 * 2; // 77.8 KB
    float* relbias      = (float*)(ws + off);          off += ((size_t)N_REL2 * 50 * 4 + 255) & ~(size_t)255;
    float* e2           = (float*)(ws + off);          off += ((size_t)N_REL2 * 50 * 4 + 255) & ~(size_t)255;

    // zero accumulators (ws is poisoned 0xAA by the harness)
    hipMemsetAsync(agg, 0, (size_t)N_NODES * 50 * 4, stream);
    hipMemsetAsync(cnt, 0, (size_t)N_NODES * 4, stream);

    // in-degree counts (same graph both layers)
    count_kernel<<<(N_EDGES + 255) / 256, 256, 0, stream>>>(dst, cnt);

    // ---- layer 1 : K=100, Kp=128, KT=4 ----
    relbias1_kernel<<<N_REL2, 64, 0, stream>>>(rel_emb, comp1, basis1, relbias);
    bcat_sw_kernel<<<NTILES * 4, 32, 0, stream>>>(basis1, root1, Bsw, D_IN, 4);
    a2bf_kernel<<<(N_NODES * 128 + 255) / 256, 256, 0, stream>>>(x, Abf, D_IN, 128,
                                                                 N_NODES * 128, tgt);
    gemm_wmma<4><<<dim3(N_NODES / 16, NTILES), 32, 0, stream>>>(Abf, Bsw, Y);
    edge_kernel<<<N_EDGES / 4, 256, 0, stream>>>(Y, src, dst, etype, comp1, relbias, agg);
    h_kernel<<<(N_NODES * 50 + 255) / 256, 256, 0, stream>>>(agg, cnt, Y, bias1, h);

    // ---- layer 2 : K=50, Kp=64, KT=2 ----
    hipMemsetAsync(agg, 0, (size_t)N_NODES * 50 * 4, stream);
    e2_kernel<<<N_REL2, 64, 0, stream>>>(rel_emb, relw, e2);
    relbias2_kernel<<<N_REL2, 64, 0, stream>>>(e2, comp2, basis2, relbias);
    bcat_sw_kernel<<<NTILES * 2, 32, 0, stream>>>(basis2, root2, Bsw, D_HID, 2);
    a2bf_kernel<<<(N_NODES * 64 + 255) / 256, 256, 0, stream>>>(h, Abf, D_HID, 64,
                                                                N_NODES * 64, nullptr);
    gemm_wmma<2><<<dim3(N_NODES / 16, NTILES), 32, 0, stream>>>(Abf, Bsw, Y);
    edge_kernel<<<N_EDGES / 4, 256, 0, stream>>>(Y, src, dst, etype, comp2, relbias, agg);

    // ---- log_softmax ----
    final_kernel<<<N_NODES / 8, 256, 0, stream>>>(agg, cnt, Y, bias2, out);
}
